// Ensemble_SRN_19782619365570
// MI455X (gfx1250) — matrix operand order repaired and checked by reading, not compiled, not run on hardware
//
#include <hip/hip_runtime.h>
#include <hip/hip_bf16.h>

typedef __attribute__((ext_vector_type(16))) _Float16 v16h;
typedef __attribute__((ext_vector_type(2)))  __fp16   v2fp16;
typedef __attribute__((ext_vector_type(8)))  float    v8f;

#define NCELLS 64
#define WPC 8            // workgroups per cell
#define WAVES_PER_WG 8   // 256 threads

// ---------------- grid index (must match reference) ----------------
__device__ __forceinline__ int cell_of(float x0, float x1, float x2) {
    int i0 = (int)((x0 + 1.0f) * 2.0f); i0 = min(max(i0, 0), 3);
    int i1 = (int)((x1 + 1.0f) * 2.0f); i1 = min(max(i1, 0), 3);
    int i2 = (int)((x2 + 1.0f) * 2.0f); i2 = min(max(i2, 0), 3);
    return i2 + 4 * i1 + 16 * i0;   // ij[:, ::-1] then base-4 little-endian
}

// ---------------- binning kernels ----------------
__global__ void k_zero(int* counts) { counts[threadIdx.x] = 0; }

__global__ void k_count(const float* __restrict__ x, int n, int* __restrict__ counts) {
    int p = blockIdx.x * blockDim.x + threadIdx.x;
    if (p < n) {
        int c = cell_of(x[3*p], x[3*p+1], x[3*p+2]);
        atomicAdd(&counts[c], 1);
    }
}

__global__ void k_scan(const int* __restrict__ counts, int* __restrict__ offsets,
                       int* __restrict__ cursor) {
    if (threadIdx.x == 0) {
        int s = 0;
        for (int m = 0; m < NCELLS; ++m) { offsets[m] = s; cursor[m] = s; s += counts[m]; }
        offsets[NCELLS] = s;
    }
}

__global__ void k_scatter(const float* __restrict__ x, int n,
                          int* __restrict__ cursor, int* __restrict__ order) {
    int p = blockIdx.x * blockDim.x + threadIdx.x;
    if (p < n) {
        int c = cell_of(x[3*p], x[3*p+1], x[3*p+2]);
        int pos = atomicAdd(&cursor[c], 1);
        order[pos] = p;
    }
}

// ---------------- hidden-unit permutation ----------------
// Slot s (C-layout storage position: slot = 16*tile + vgpr + 8*hiHalf) holds
// hidden unit perm(s). Chosen so the f32 C layout of one layer IS the f16 B
// layout of the next after pure in-lane cvt_pk packing (no shuffles/selects):
// swap the middle two 8-blocks of each 32-block.
__device__ __forceinline__ int perm(int s) {
    int r   = s & 31;
    int blk = r >> 3;                       // 0..3
    int nb  = (blk == 1) ? 2 : (blk == 2 ? 1 : blk);
    return (s & ~31) + (nb << 3) + (r & 7);
}

// ---------------- f16 pack helpers ----------------
__device__ __forceinline__ unsigned pk2(float lo, float hi) {
    union { v2fp16 h; unsigned u; } v;
    v.h = __builtin_amdgcn_cvt_pkrtz(lo, hi);   // single v_cvt_pk_rtz_f16_f32
    return v.u;
}

// With the slot permutation, C->B conversion is pure in-lane packing:
//   B[j]   = pack(ce[2j], ce[2j+1])   (lo lanes: K 0-7,  hi lanes: K 16-23)
//   B[4+j] = pack(co[2j], co[2j+1])   (lo lanes: K 8-15, hi lanes: K 24-31)
__device__ __forceinline__ v16h packB(const v8f& ce, const v8f& co) {
    union { unsigned u[8]; v16h h; } r;
#pragma unroll
    for (int j = 0; j < 4; ++j) r.u[j]     = pk2(ce[2*j], ce[2*j+1]);
#pragma unroll
    for (int j = 0; j < 4; ++j) r.u[4 + j] = pk2(co[2*j], co[2*j+1]);
    return r.h;
}

// single-instruction relu: v_med3_f32(x, 0, +FLT_MAX)
__device__ __forceinline__ float relu(float x) {
    return __builtin_amdgcn_fmed3f(x, 0.0f, 3.402823466e38f);
}

#define WMMA(A, B, C) __builtin_amdgcn_wmma_f32_16x16x32_f16(false, (A), false, (B), (short)0, (C), false, false)

// ---------------- fused MLP kernel (weights stationary as A) ----------------
__global__ void __launch_bounds__(256)
k_mlp(const float* __restrict__ X,
      const float* __restrict__ W1, const float* __restrict__ B1,
      const float* __restrict__ W2, const float* __restrict__ B2,
      const float* __restrict__ W3, const float* __restrict__ B3,
      const int* __restrict__ offsets, const int* __restrict__ order,
      float* __restrict__ out) {
    const int m    = blockIdx.x / WPC;
    const int lane = threadIdx.x & 31;
    const int wave = (threadIdx.x >> 5) + (blockIdx.x % WPC) * WAVES_PER_WG; // 0..63 within cell
    const bool hi  = lane >= 16;
    const int  ln  = lane & 15;

    const int start = offsets[m];
    const int cnt   = offsets[m + 1] - start;
    if (cnt == 0) return;

    const float* w1 = W1 + m * 3 * 64;
    const float* b1 = B1 + m * 64;
    const float* w2 = W2 + m * 64 * 64;
    const float* b2 = B2 + m * 64;
    const float* w3 = W3 + m * 64;
    const float  b3 = B3[m];

    // ---- A1: layer1 weights (16x32 f16). Row M of tile t computes hidden
    //      unit perm(16t+M). K=0..2 = W1 rows, K=3 = bias row (pairs with 1.0).
    v16h a1[4];
#pragma unroll
    for (int i = 0; i < 4; ++i) {
        v16h a = {};
        if (!hi) {                       // lanes<16 hold K0-7; only K0..3 nonzero
            int n = perm(16 * i + ln);
            a[0] = (_Float16)w1[0 * 64 + n];
            a[1] = (_Float16)w1[1 * 64 + n];
            a[2] = (_Float16)w1[2 * 64 + n];
            a[3] = (_Float16)b1[n];
        }
        a1[i] = a;
    }

    // ---- A2: layer2 weights. Row -> output unit perm(16i+ln); K-side indexes
    //      input units in NATURAL order (B supplies units in natural K order).
    v16h a2[4][2];
#pragma unroll
    for (int i = 0; i < 4; ++i)
#pragma unroll
        for (int kt = 0; kt < 2; ++kt) {
            v16h a = {};
            int n  = perm(16 * i + ln);
            int kb = 32 * kt + (hi ? 16 : 0);
#pragma unroll
            for (int j = 0; j < 4; ++j) {
                a[2*j]     = (_Float16)w2[(kb + 2*j)     * 64 + n];
                a[2*j + 1] = (_Float16)w2[(kb + 2*j + 1) * 64 + n];
            }
            int kc = kb + 8;
#pragma unroll
            for (int j = 0; j < 4; ++j) {
                a[8 + 2*j]     = (_Float16)w2[(kc + 2*j)     * 64 + n];
                a[8 + 2*j + 1] = (_Float16)w2[(kc + 2*j + 1) * 64 + n];
            }
            a2[i][kt] = a;
        }

    // ---- A3: layer3 weights as single row M=0, natural K order ----
    v16h a3[2];
#pragma unroll
    for (int kt = 0; kt < 2; ++kt) {
        v16h a = {};
        if (ln == 0) {
            int kb = 32 * kt + (hi ? 16 : 0);
#pragma unroll
            for (int j = 0; j < 4; ++j) {
                a[2*j]     = (_Float16)w3[kb + 2*j];
                a[2*j + 1] = (_Float16)w3[kb + 2*j + 1];
            }
            int kc = kb + 8;
#pragma unroll
            for (int j = 0; j < 4; ++j) {
                a[8 + 2*j]     = (_Float16)w3[kc + 2*j];
                a[8 + 2*j + 1] = (_Float16)w3[kc + 2*j + 1];
            }
        }
        a3[kt] = a;
    }

    // ---- layer-2 bias preloaded into accumulator-layout registers (permuted slots) ----
    v8f c2i[4];
#pragma unroll
    for (int i = 0; i < 4; ++i)
#pragma unroll
        for (int r = 0; r < 8; ++r)
            c2i[i][r] = b2[perm(16 * i + r + (hi ? 8 : 0))];

    const int ntiles = (cnt + 15) >> 4;
    for (int t = wave; t < ntiles; t += NCELLS) {
        // ---- gather 16 points (clamped; masking via select keeps EXEC full) ----
        int ii     = t * 16 + ln;
        bool valid = ii < cnt;
        int pid    = order[start + (valid ? ii : 0)];
        float x0 = X[3*pid], x1 = X[3*pid + 1], x2 = X[3*pid + 2];

        v16h bx = {};
        if (!hi) {                       // lanes<16 hold K0-15; K0..2 = x, K3 = 1 (bias)
            bx[0] = (_Float16)x0;
            bx[1] = (_Float16)x1;
            bx[2] = (_Float16)x2;
            bx[3] = (_Float16)1.0f;
        }

        // ---- layer 1: H1 = relu(W1^T X^T + b1), outputs in permuted slots ----
        v8f c1[4];
#pragma unroll
        for (int i = 0; i < 4; ++i) {
            v8f z = {};
            c1[i] = WMMA(a1[i], bx, z);
#pragma unroll
            for (int r = 0; r < 8; ++r) c1[i][r] = relu(c1[i][r]);
        }

        // ---- layer 2: H2 = relu(W2^T H1 + b2) ----
        v16h bh0 = packB(c1[0], c1[1]);   // units 0-31 in natural K order
        v16h bh1 = packB(c1[2], c1[3]);   // units 32-63
        v8f c2[4];
#pragma unroll
        for (int i = 0; i < 4; ++i) {
            v8f c = c2i[i];
            c = WMMA(a2[i][0], bh0, c);
            c = WMMA(a2[i][1], bh1, c);
#pragma unroll
            for (int r = 0; r < 8; ++r) c[r] = relu(c[r]);
            c2[i] = c;
        }

        // ---- layer 3: y = W3^T H2 + b3 (row 0 of D) ----
        v16h bg0 = packB(c2[0], c2[1]);
        v16h bg1 = packB(c2[2], c2[3]);
        v8f c3 = {};
        c3 = WMMA(a3[0], bg0, c3);
        c3 = WMMA(a3[1], bg1, c3);

        if (!hi && valid) out[pid] = c3[0] + b3;   // VGPR0 lanes0-15: M=0, N=point
    }
}

// ---------------- launcher ----------------
extern "C" void kernel_launch(void* const* d_in, const int* in_sizes, int n_in,
                              void* d_out, int out_size, void* d_ws, size_t ws_size,
                              hipStream_t stream) {
    const float* x  = (const float*)d_in[0];
    const float* W1 = (const float*)d_in[1];
    const float* B1 = (const float*)d_in[2];
    const float* W2 = (const float*)d_in[3];
    const float* B2 = (const float*)d_in[4];
    const float* W3 = (const float*)d_in[5];
    const float* B3 = (const float*)d_in[6];
    float* out = (float*)d_out;

    const int N = in_sizes[0] / 3;

    char* ws = (char*)d_ws;
    int* counts  = (int*)(ws);              // 64 ints
    int* offsets = (int*)(ws + 256);        // 65 ints
    int* cursor  = (int*)(ws + 512);        // 64 ints
    int* order   = (int*)(ws + 1024);       // N ints

    const int blk = 256;
    const int nb  = (N + blk - 1) / blk;

    k_zero   <<<1, NCELLS, 0, stream>>>(counts);
    k_count  <<<nb, blk, 0, stream>>>(x, N, counts);
    k_scan   <<<1, 1, 0, stream>>>(counts, offsets, cursor);
    k_scatter<<<nb, blk, 0, stream>>>(x, N, cursor, order);
    k_mlp    <<<NCELLS * WPC, 256, 0, stream>>>(x, W1, B1, W2, B2, W3, B3,
                                                offsets, order, out);
}
